// DAWNBlock_76742475645442
// MI455X (gfx1250) — compile-verified, matrix operand order checked by reading, not run on hardware
//
#include <hip/hip_runtime.h>
#include <cstdint>

// ---------------- constants ----------------
#define BB 8
#define SS 1024
#define DD 1024
#define RR 128
#define NH 16
#define DH 64
#define NCOMP 64
#define NEXP 32
#define NKNOW 16384
#define TOPC 8
#define TOPE 4
#define KKNOW 8

typedef __attribute__((ext_vector_type(16))) __bf16 v16bf;
typedef __attribute__((ext_vector_type(8)))  float  v8f;
typedef __attribute__((ext_vector_type(4)))  unsigned int v4u;
typedef __attribute__((ext_vector_type(8)))  int    v8i;
typedef __attribute__((ext_vector_type(4)))  int    v4i;

#define DEV __device__ __forceinline__

DEV v8f wmma_bf16(v16bf a, v16bf b, v8f c) {
  return __builtin_amdgcn_wmma_f32_16x16x32_bf16(false, a, false, b, (short)0, c,
                                                 false, false);
}

// A-fragment / Bt-fragment loader (16-bit, 16x32 layout per ISA 7.12.2):
// lane L: row = row0 + (L&15), half = L>>4; VGPR v holds K = k0 + 2(v&3) + 8*half + 16*(v>>2), +1
DEV v16bf load_frag(const __bf16* base, int ld, int row, int half, int k0) {
  v16bf f;
#pragma unroll
  for (int v = 0; v < 8; ++v) {
    int kb = k0 + ((v & 3) << 1) + (half << 3) + ((v >> 2) << 4);
    const __bf16* p = base + (long)row * ld + kb;
    f[2 * v]     = p[0];
    f[2 * v + 1] = p[1];
  }
  return f;
}

DEV uint32_t lds_off(const void* p) { return (uint32_t)(uintptr_t)p; }

// ---- Tensor Data Mover: 2D bf16 tile (w elems x h rows, row stride in elems) -> LDS ----
// D# per CDNA5 ISA ch.8: group0 = {count/flags, lds_addr, global_addr[56:0], type=2},
// group1 = {data_size=2B, tensor_dim0/1, tile_dim0/1, tensor_dim0_stride}.
DEV void tdm_load_2d_bf16(const void* gptr, uint32_t lds, uint32_t w, uint32_t h,
                          uint32_t stride) {
  uint64_t ga = (uint64_t)(uintptr_t)gptr;
  v4u g0;
  g0[0] = 1u;                                            // count=1, user descriptor
  g0[1] = lds;                                           // lds_addr
  g0[2] = (uint32_t)ga;                                  // global_addr[31:0]
  g0[3] = (uint32_t)((ga >> 32) & 0x01FFFFFFu) | (2u << 30);  // addr[56:32] | type=2
  v8i g1;
  g1[0] = (int)(1u << 16);                               // data_size = 1 (2 bytes)
  g1[1] = (int)((w & 0xFFFFu) << 16);                    // tensor_dim0 lo -> [31:16]
  g1[2] = (int)(((w >> 16) & 0xFFFFu) | ((h & 0xFFFFu) << 16));  // dim0 hi | tensor_dim1 lo
  g1[3] = (int)(((h >> 16) & 0xFFFFu) | ((w & 0xFFFFu) << 16));  // dim1 hi | tile_dim0
  g1[4] = (int)(h & 0xFFFFu);                            // tile_dim1 (tile_dim2 = 0)
  g1[5] = (int)stride;                                   // tensor_dim0_stride lo32
  g1[6] = 0;                                             // stride hi | dim1_stride lo
  g1[7] = 0;
  v4i g2 = {0, 0, 0, 0}, g3 = {0, 0, 0, 0};
#if __clang_major__ >= 23
  v8i gx = {0, 0, 0, 0, 0, 0, 0, 0};
  __builtin_amdgcn_tensor_load_to_lds(g0, g1, g2, g3, gx, 0);
#else
  __builtin_amdgcn_tensor_load_to_lds(g0, g1, g2, g3, 0);
#endif
}

// ---------------- LayerNorm: f32 in -> bf16 out ----------------
__global__ void k_ln(const float* __restrict__ x, const float* __restrict__ sc,
                     const float* __restrict__ bi, __bf16* __restrict__ out) {
  int row = blockIdx.x, t = threadIdx.x;
  const float* xr = x + (long)row * DD;
  __shared__ float red[256];
  float ls = 0.f;
  for (int i = t; i < DD; i += 256) ls += xr[i];
  red[t] = ls; __syncthreads();
  for (int o = 128; o > 0; o >>= 1) { if (t < o) red[t] += red[t + o]; __syncthreads(); }
  float mu = red[0] / DD; __syncthreads();
  float lv = 0.f;
  for (int i = t; i < DD; i += 256) { float d = xr[i] - mu; lv += d * d; }
  red[t] = lv; __syncthreads();
  for (int o = 128; o > 0; o >>= 1) { if (t < o) red[t] += red[t + o]; __syncthreads(); }
  float rstd = rsqrtf(red[0] / DD + 1e-5f);
  for (int i = t; i < DD; i += 256)
    out[(long)row * DD + i] = (__bf16)((xr[i] - mu) * rstd * sc[i] + bi[i]);
}

__global__ void k_zero(float* p, int n) {
  int t = blockIdx.x * 256 + threadIdx.x;
  if (t < n) p[t] = 0.f;
}

// ---------------- router dense accumulation ----------------
__global__ void k_route(const __bf16* __restrict__ nx, const float* __restrict__ imp,
                        const float* __restrict__ W, int nR, float* __restrict__ dense) {
  int bs = blockIdx.x, b = bs / SS, t = threadIdx.x;  // blockDim = 64
  __shared__ float rw[DD];
  __shared__ float red[64];
  const __bf16* xr = nx + (long)bs * DD;
  for (int i = t; i < DD; i += 64) rw[i] = (float)xr[i];
  __syncthreads();
  float logit = -3.0e38f;
  if (t < nR) {
    float a = 0.f;
    for (int d = 0; d < DD; ++d) a += rw[d] * W[d * nR + t];
    logit = a;
  }
  red[t] = logit; __syncthreads();
  for (int o = 32; o > 0; o >>= 1) { if (t < o) red[t] = fmaxf(red[t], red[t + o]); __syncthreads(); }
  float mx = red[0]; __syncthreads();
  float e = (t < nR) ? __expf(logit - mx) : 0.f;
  red[t] = e; __syncthreads();
  for (int o = 32; o > 0; o >>= 1) { if (t < o) red[t] += red[t + o]; __syncthreads(); }
  float sm = red[0];
  if (t < nR) atomicAdd(&dense[b * nR + t], imp[bs] * e / sm);
}

// ---------------- top-k sparsify (compacted) ----------------
__global__ void k_topk(const float* __restrict__ dense, int nR, int k,
                       int* __restrict__ oi, float* __restrict__ ow) {
  if (threadIdx.x != 0) return;
  int b = blockIdx.x;
  float v[64], tv[8];
  for (int i = 0; i < nR; ++i) v[i] = dense[b * nR + i];
  float ssum = 0.f;
  for (int j = 0; j < k; ++j) {
    float best = -3.0e38f; int bi = 0;
    for (int i = 0; i < nR; ++i) if (v[i] > best) { best = v[i]; bi = i; }
    oi[b * k + j] = bi; tv[j] = best; v[bi] = -3.0e38f; ssum += best;
  }
  for (int j = 0; j < k; ++j) ow[b * k + j] = tv[j] / (ssum + 1e-8f);
}

// ---------------- synthesize mixed matrices (store transposed, bf16) ----------------
__global__ void k_synth_sc(const int* __restrict__ oi, const float* __restrict__ ow,
                           const float* __restrict__ cn, __bf16* __restrict__ scT, int k) {
  long t = (long)blockIdx.x * 256 + threadIdx.x;
  long total = (long)BB * RR * DD;
  if (t >= total) return;
  int b = (int)(t / (RR * DD)); int rd = (int)(t % (RR * DD));
  int r = rd / DD, d = rd % DD;
  float a = 0.f;
  for (int j = 0; j < k; ++j)
    a += ow[b * k + j] * cn[(long)oi[b * k + j] * DD * RR + (long)d * RR + r];
  scT[t] = (__bf16)a;
}
__global__ void k_synth_e(const int* __restrict__ oi, const float* __restrict__ ow,
                          const float* __restrict__ pool, __bf16* __restrict__ eT, int k) {
  long t = (long)blockIdx.x * 256 + threadIdx.x;
  long total = (long)BB * DD * RR;
  if (t >= total) return;
  int b = (int)(t / (DD * RR)); int dr = (int)(t % (DD * RR));
  int d = dr / RR, r = dr % RR;
  float a = 0.f;
  for (int j = 0; j < k; ++j)
    a += ow[b * k + j] * pool[(long)oi[b * k + j] * RR * DD + (long)r * DD + d];
  eT[t] = (__bf16)a;
}

// ---------------- conversions ----------------
__global__ void k_conv(const float* __restrict__ in, __bf16* __restrict__ out, long n) {
  long t = (long)blockIdx.x * 256 + threadIdx.x;
  if (t < n) out[t] = (__bf16)in[t];
}
__global__ void k_conv_t(const float* __restrict__ W, __bf16* __restrict__ Wt) {
  long t = (long)blockIdx.x * 256 + threadIdx.x;
  if (t >= (long)DD * DD) return;
  int n = (int)(t / DD), kk = (int)(t % DD);
  Wt[t] = (__bf16)W[(long)kk * DD + n];
}

// ---------------- batched bf16 GEMM with TDM-staged B panel ----------------
// C = A[MxK] * Bt[NxK]^T ; Bt tile (64 rows x 32 k) double-buffered through LDS via TDM.
__global__ void k_gemm_bf16(const __bf16* __restrict__ A, long sa,
                            const __bf16* __restrict__ Bt, long sb,
                            __bf16* __restrict__ C, long sc, int N, int K) {
  __shared__ __bf16 tB[2][64 * 32];
  int lane = threadIdx.x, half = lane >> 4, ln = lane & 15;
  const __bf16* Ab = A + (long)blockIdx.x * sa;
  const __bf16* Bb = Bt + (long)blockIdx.x * sb;
  __bf16* Cb = C + (long)blockIdx.x * sc;
  int row0 = blockIdx.y * 16, col0 = blockIdx.z * 64;
  v8f acc[4]; v8f z = {};
#pragma unroll
  for (int g = 0; g < 4; ++g) acc[g] = z;
  tdm_load_2d_bf16(Bb + (long)col0 * K, lds_off(&tB[0][0]), 32, 64, (uint32_t)K);
  for (int k0 = 0; k0 < K; k0 += 32) {
    int buf = (k0 >> 5) & 1;
    if (k0 + 32 < K) {
      tdm_load_2d_bf16(Bb + (long)col0 * K + (k0 + 32), lds_off(&tB[buf ^ 1][0]), 32, 64,
                       (uint32_t)K);
      __builtin_amdgcn_s_wait_tensorcnt(1);
    } else {
      __builtin_amdgcn_s_wait_tensorcnt(0);
    }
    asm volatile("" ::: "memory");
    if (k0 + 32 < K)
      __builtin_prefetch(Ab + (long)(row0 + ln) * K + k0 + 32, 0, 0);
    v16bf va = load_frag(Ab, K, row0 + ln, half, k0);
#pragma unroll
    for (int g = 0; g < 4; ++g) {
      v16bf vb = load_frag(&tB[buf][0], 32, 16 * g + ln, half, 0);
      acc[g] = wmma_bf16(va, vb, acc[g]);
    }
  }
#pragma unroll
  for (int g = 0; g < 4; ++g)
#pragma unroll
    for (int r = 0; r < 8; ++r) {
      int m = r + (half ? 8 : 0);
      Cb[(long)(row0 + m) * N + col0 + 16 * g + ln] = (__bf16)acc[g][r];
    }
}

// ---------------- W_O GEMM + residual -> f32 (TDM-staged B panel) ----------------
__global__ void k_gemm_wo(const __bf16* __restrict__ A, const __bf16* __restrict__ Bt,
                          const float* __restrict__ xres, float* __restrict__ C) {
  __shared__ __bf16 tB[2][64 * 32];
  int lane = threadIdx.x, half = lane >> 4, ln = lane & 15;
  int row0 = blockIdx.x * 16, col0 = blockIdx.y * 64;
  const int K = DD, N = DD;
  v8f acc[4]; v8f z = {};
#pragma unroll
  for (int g = 0; g < 4; ++g) acc[g] = z;
  tdm_load_2d_bf16(Bt + (long)col0 * K, lds_off(&tB[0][0]), 32, 64, (uint32_t)K);
  for (int k0 = 0; k0 < K; k0 += 32) {
    int buf = (k0 >> 5) & 1;
    if (k0 + 32 < K) {
      tdm_load_2d_bf16(Bt + (long)col0 * K + (k0 + 32), lds_off(&tB[buf ^ 1][0]), 32, 64,
                       (uint32_t)K);
      __builtin_amdgcn_s_wait_tensorcnt(1);
    } else {
      __builtin_amdgcn_s_wait_tensorcnt(0);
    }
    asm volatile("" ::: "memory");
    if (k0 + 32 < K)
      __builtin_prefetch(A + (long)(row0 + ln) * K + k0 + 32, 0, 0);
    v16bf va = load_frag(A, K, row0 + ln, half, k0);
#pragma unroll
    for (int g = 0; g < 4; ++g) {
      v16bf vb = load_frag(&tB[buf][0], 32, 16 * g + ln, half, 0);
      acc[g] = wmma_bf16(va, vb, acc[g]);
    }
  }
#pragma unroll
  for (int g = 0; g < 4; ++g)
#pragma unroll
    for (int r = 0; r < 8; ++r) {
      int m = r + (half ? 8 : 0);
      long idx = (long)(row0 + m) * N + col0 + 16 * g + ln;
      C[idx] = acc[g][r] + xres[idx];
    }
}

// ---------------- flash causal attention: one wave = 16 queries, one (b,h) ----------------
__global__ void k_attn(const __bf16* __restrict__ Q, const __bf16* __restrict__ Kt,
                       const __bf16* __restrict__ V, __bf16* __restrict__ O) {
  int lane = threadIdx.x, half = lane >> 4, ln = lane & 15;
  int bh = blockIdx.x, b = bh >> 4, h = bh & 15;
  int q0 = blockIdx.y * 16;
  const __bf16* Qb = Q + (long)b * SS * DD + h * DH;
  const __bf16* Kb = Kt + (long)b * SS * DD + h * DH;
  const __bf16* Vb = V + (long)b * SS * DD + h * DH;
  __bf16* Ob = O + (long)b * SS * DD + h * DH;

  v16bf aq0 = load_frag(Qb, DD, q0 + ln, half, 0);
  v16bf aq1 = load_frag(Qb, DD, q0 + ln, half, 32);
  v8f o[4]; v8f z = {};
#pragma unroll
  for (int g = 0; g < 4; ++g) o[g] = z;

  __shared__ float sP[16][32];
  __shared__ float sM[16], sL[16], sC[16];
  if (lane < 16) { sM[lane] = -3.0e38f; sL[lane] = 0.f; }
  __syncthreads();

  for (int j0 = 0; j0 <= q0 + 15; j0 += 32) {
#pragma unroll
    for (int sub = 0; sub < 2; ++sub) {
      int st = j0 + 16 * sub;
      if (st <= q0 + 15) {
        v16bf bk0 = load_frag(Kb, DD, st + ln, half, 0);
        v16bf bk1 = load_frag(Kb, DD, st + ln, half, 32);
        v8f s = z;
        s = wmma_bf16(aq0, bk0, s);
        s = wmma_bf16(aq1, bk1, s);
#pragma unroll
        for (int r = 0; r < 8; ++r) {
          int m = r + (half ? 8 : 0);
          int key = st + ln;
          float val = s[r] * 0.125f;
          if (key > q0 + m) val = -3.0e38f;
          sP[m][16 * sub + ln] = val;
        }
      } else {
#pragma unroll
        for (int r = 0; r < 8; ++r) sP[r + (half ? 8 : 0)][16 * sub + ln] = -3.0e38f;
      }
    }
    __syncthreads();
    if (lane < 16) {
      float mx = sM[lane];
      for (int j = 0; j < 32; ++j) mx = fmaxf(mx, sP[lane][j]);
      float corr = __expf(sM[lane] - mx);
      float ps = 0.f;
      for (int j = 0; j < 32; ++j) { float p = __expf(sP[lane][j] - mx); sP[lane][j] = p; ps += p; }
      sL[lane] = sL[lane] * corr + ps;
      sM[lane] = mx; sC[lane] = corr;
    }
    __syncthreads();
#pragma unroll
    for (int g = 0; g < 4; ++g)
#pragma unroll
      for (int r = 0; r < 8; ++r) o[g][r] *= sC[r + (half ? 8 : 0)];
    v16bf ap;
#pragma unroll
    for (int v = 0; v < 8; ++v) {
      int kb = ((v & 3) << 1) + (half << 3) + ((v >> 2) << 4);
      ap[2 * v]     = (__bf16)sP[ln][kb];
      ap[2 * v + 1] = (__bf16)sP[ln][kb + 1];
    }
#pragma unroll
    for (int g = 0; g < 4; ++g) {
      v16bf bv;
#pragma unroll
      for (int v = 0; v < 8; ++v) {
        int kb = ((v & 3) << 1) + (half << 3) + ((v >> 2) << 4);
        int r0 = j0 + kb;     if (r0 > SS - 1) r0 = SS - 1;
        int r1 = j0 + kb + 1; if (r1 > SS - 1) r1 = SS - 1;
        bv[2 * v]     = Vb[(long)r0 * DD + 16 * g + ln];
        bv[2 * v + 1] = Vb[(long)r1 * DD + 16 * g + ln];
      }
      o[g] = wmma_bf16(ap, bv, o[g]);
    }
    __syncthreads();
  }
#pragma unroll
  for (int g = 0; g < 4; ++g)
#pragma unroll
    for (int r = 0; r < 8; ++r) {
      int m = r + (half ? 8 : 0);
      Ob[(long)(q0 + m) * DD + 16 * g + ln] = (__bf16)(o[g][r] / sL[m]);
    }
}

// ---------------- knowledge scoring (TDM-staged keys) + top-8 + gather ----------------
__global__ void k_mem(const __bf16* __restrict__ Qm, const __bf16* __restrict__ Kb,
                      const float* __restrict__ Vk, const float* __restrict__ x2,
                      float* __restrict__ out) {
  __shared__ __bf16 tK[2][16 * 128];
  int lane = threadIdx.x, half = lane >> 4, ln = lane & 15;
  int row0 = blockIdx.x * 16;
  v16bf a[4];
#pragma unroll
  for (int t = 0; t < 4; ++t) a[t] = load_frag(Qm, RR, row0 + ln, half, 32 * t);

  __shared__ float sS[16][16];
  __shared__ float sW[16][KKNOW];
  __shared__ int   sI[16][KKNOW];
  float tv[KKNOW]; int ti[KKNOW];
  if (lane < 16)
#pragma unroll
    for (int j = 0; j < KKNOW; ++j) { tv[j] = -3.0e38f; ti[j] = 0; }

  const int NT = NKNOW / 16;
  tdm_load_2d_bf16(Kb, lds_off(&tK[0][0]), 128, 16, 128);
  for (int kt = 0; kt < NT; ++kt) {
    int buf = kt & 1;
    if (kt + 1 < NT) {
      tdm_load_2d_bf16(Kb + (long)(kt + 1) * 16 * RR, lds_off(&tK[buf ^ 1][0]), 128, 16, 128);
      __builtin_amdgcn_s_wait_tensorcnt(1);
    } else {
      __builtin_amdgcn_s_wait_tensorcnt(0);
    }
    asm volatile("" ::: "memory");
    v8f acc = {};
#pragma unroll
    for (int t = 0; t < 4; ++t) {
      v16bf vb = load_frag(&tK[buf][0], RR, ln, half, 32 * t);
      acc = wmma_bf16(a[t], vb, acc);
    }
#pragma unroll
    for (int r = 0; r < 8; ++r)
      sS[r + (half ? 8 : 0)][ln] = acc[r] * 0.0883883476f;  // 1/sqrt(128)
    __syncthreads();
    if (lane < 16) {
      int col0 = kt * 16;
      for (int j = 0; j < 16; ++j) {
        float v = sS[lane][j];
        int mi = 0; float mv = tv[0];
#pragma unroll
        for (int q = 1; q < KKNOW; ++q) if (tv[q] < mv) { mv = tv[q]; mi = q; }
        if (v > mv) { tv[mi] = v; ti[mi] = col0 + j; }
      }
    }
    __syncthreads();
  }
  if (lane < 16) {
    float mx = -3.0e38f;
#pragma unroll
    for (int j = 0; j < KKNOW; ++j) mx = fmaxf(mx, tv[j]);
    float sm = 0.f; float e[KKNOW];
#pragma unroll
    for (int j = 0; j < KKNOW; ++j) { e[j] = __expf(tv[j] - mx); sm += e[j]; }
#pragma unroll
    for (int j = 0; j < KKNOW; ++j) { sW[lane][j] = e[j] / sm; sI[lane][j] = ti[j]; }
  }
  __syncthreads();
  for (int m = 0; m < 16; ++m) {
    long grow = row0 + m;
    for (int d = lane; d < DD; d += 32) {
      float accv = x2[grow * DD + d];
#pragma unroll
      for (int j = 0; j < KKNOW; ++j)
        accv += sW[m][j] * Vk[(long)sI[m][j] * DD + d];
      out[grow * DD + d] = accv;
    }
  }
}

// ---------------- launch ----------------
extern "C" void kernel_launch(void* const* d_in, const int* in_sizes, int n_in,
                              void* d_out, int out_size, void* d_ws, size_t ws_size,
                              hipStream_t stream) {
  const float* x    = (const float*)d_in[0];
  const float* imp  = (const float*)d_in[1];
  const float* Wc   = (const float*)d_in[2];
  const float* WQr  = (const float*)d_in[3];
  const float* WKr  = (const float*)d_in[4];
  const float* WVr  = (const float*)d_in[5];
  const float* Wm   = (const float*)d_in[6];
  const float* cn   = (const float*)d_in[7];
  const float* pool = (const float*)d_in[8];
  const float* knowK = (const float*)d_in[9];
  const float* knowV = (const float*)d_in[10];
  const float* W_O  = (const float*)d_in[11];
  const float* ln1s = (const float*)d_in[12];
  const float* ln1b = (const float*)d_in[13];
  const float* ln2s = (const float*)d_in[14];
  const float* ln2b = (const float*)d_in[15];
  float* out = (float*)d_out;

  char* ws = (char*)d_ws;
  size_t off = 0;
  auto carve = [&](size_t bytes) -> void* {
    void* p = ws + off;
    off = (off + bytes + 255) & ~(size_t)255;
    return p;
  };
  const long BSD = (long)BB * SS * DD;
  const long BSR = (long)BB * SS * RR;
  const long BRD = (long)BB * RR * DD;

  __bf16* n1bf  = (__bf16*)carve(BSD * 2);
  __bf16* n2bf  = (__bf16*)carve(BSD * 2);
  __bf16* Qbf   = (__bf16*)carve(BSD * 2);
  __bf16* Kbf   = (__bf16*)carve(BSD * 2);
  __bf16* Vbf   = (__bf16*)carve(BSD * 2);
  __bf16* attnb = (__bf16*)carve(BSD * 2);
  __bf16* hbf   = (__bf16*)carve(BSR * 2);
  __bf16* Qmbf  = (__bf16*)carve(BSR * 2);
  __bf16* scT   = (__bf16*)carve(BRD * 2);
  __bf16* scmT  = (__bf16*)carve(BRD * 2);
  __bf16* eqT   = (__bf16*)carve(BRD * 2);
  __bf16* ekT   = (__bf16*)carve(BRD * 2);
  __bf16* evT   = (__bf16*)carve(BRD * 2);
  __bf16* WOt   = (__bf16*)carve((long)DD * DD * 2);
  __bf16* knowKb = (__bf16*)carve((long)NKNOW * RR * 2);
  float*  x2    = (float*)carve(BSD * 4);
  float*  dense = (float*)carve(2048 * 4);
  int*    tki   = (int*)carve(512 * 4);
  float*  tkv   = (float*)carve(512 * 4);

  float* cw_d = dense;         float* eq_d = dense + 512;
  float* ek_d = dense + 768;   float* ev_d = dense + 1024;
  float* mw_d = dense + 1280;
  int*   cw_i = tki;        float* cw_w = tkv;
  int*   eq_i = tki + 64;   float* eq_w = tkv + 64;
  int*   ek_i = tki + 96;   float* ek_w = tkv + 96;
  int*   ev_i = tki + 128;  float* ev_w = tkv + 128;
  int*   mw_i = tki + 160;  float* mw_w = tkv + 160;

  k_ln<<<BB * SS, 256, 0, stream>>>(x, ln1s, ln1b, n1bf);
  k_zero<<<(2048 + 255) / 256, 256, 0, stream>>>(dense, 2048);
  k_route<<<BB * SS, 64, 0, stream>>>(n1bf, imp, Wc, NCOMP, cw_d);
  k_route<<<BB * SS, 64, 0, stream>>>(n1bf, imp, WQr, NEXP, eq_d);
  k_route<<<BB * SS, 64, 0, stream>>>(n1bf, imp, WKr, NEXP, ek_d);
  k_route<<<BB * SS, 64, 0, stream>>>(n1bf, imp, WVr, NEXP, ev_d);
  k_topk<<<BB, 32, 0, stream>>>(cw_d, NCOMP, TOPC, cw_i, cw_w);
  k_topk<<<BB, 32, 0, stream>>>(eq_d, NEXP, TOPE, eq_i, eq_w);
  k_topk<<<BB, 32, 0, stream>>>(ek_d, NEXP, TOPE, ek_i, ek_w);
  k_topk<<<BB, 32, 0, stream>>>(ev_d, NEXP, TOPE, ev_i, ev_w);
  k_synth_sc<<<(int)((BRD + 255) / 256), 256, 0, stream>>>(cw_i, cw_w, cn, scT, TOPC);
  k_synth_e<<<(int)((BRD + 255) / 256), 256, 0, stream>>>(eq_i, eq_w, pool, eqT, TOPE);
  k_synth_e<<<(int)((BRD + 255) / 256), 256, 0, stream>>>(ek_i, ek_w, pool, ekT, TOPE);
  k_synth_e<<<(int)((BRD + 255) / 256), 256, 0, stream>>>(ev_i, ev_w, pool, evT, TOPE);
  k_conv<<<(int)(((long)NKNOW * RR + 255) / 256), 256, 0, stream>>>(knowK, knowKb,
                                                                   (long)NKNOW * RR);
  k_conv_t<<<(int)(((long)DD * DD + 255) / 256), 256, 0, stream>>>(W_O, WOt);
  // h = n1 @ shared_compress
  k_gemm_bf16<<<dim3(BB, SS / 16, RR / 64), 32, 0, stream>>>(
      n1bf, (long)SS * DD, scT, (long)RR * DD, hbf, (long)SS * RR, RR, DD);
  // Q/K/V = h @ E*
  k_gemm_bf16<<<dim3(BB, SS / 16, DD / 64), 32, 0, stream>>>(
      hbf, (long)SS * RR, eqT, (long)DD * RR, Qbf, (long)SS * DD, DD, RR);
  k_gemm_bf16<<<dim3(BB, SS / 16, DD / 64), 32, 0, stream>>>(
      hbf, (long)SS * RR, ekT, (long)DD * RR, Kbf, (long)SS * DD, DD, RR);
  k_gemm_bf16<<<dim3(BB, SS / 16, DD / 64), 32, 0, stream>>>(
      hbf, (long)SS * RR, evT, (long)DD * RR, Vbf, (long)SS * DD, DD, RR);
  k_attn<<<dim3(BB * NH, SS / 16), 32, 0, stream>>>(Qbf, Kbf, Vbf, attnb);
  k_gemm_wo<<<dim3((BB * SS) / 16, DD / 64), 32, 0, stream>>>(attnb, WOt, x, x2);
  k_ln<<<BB * SS, 256, 0, stream>>>(x2, ln2s, ln2b, n2bf);
  k_route<<<BB * SS, 64, 0, stream>>>(n2bf, imp, Wm, NCOMP, mw_d);
  k_topk<<<BB, 32, 0, stream>>>(mw_d, NCOMP, TOPC, mw_i, mw_w);
  k_synth_sc<<<(int)((BRD + 255) / 256), 256, 0, stream>>>(mw_i, mw_w, cn, scmT, TOPC);
  k_gemm_bf16<<<dim3(BB, SS / 16, RR / 64), 32, 0, stream>>>(
      n2bf, (long)SS * DD, scmT, (long)RR * DD, Qmbf, (long)SS * RR, RR, DD);
  k_mem<<<(BB * SS) / 16, 32, 0, stream>>>(Qmbf, knowKb, knowV, x2, out);

  (void)in_sizes; (void)n_in; (void)out_size; (void)ws_size;
}
